// GRec_78125455114707
// MI455X (gfx1250) — matrix operand order repaired and checked by reading, not compiled, not run on hardware
//
#include <hip/hip_runtime.h>
#include <hip/hip_bf16.h>

typedef _Float16 h16;
typedef __attribute__((ext_vector_type(16))) _Float16 v16h;
typedef __attribute__((ext_vector_type(8)))  float    v8f;

#define B_SZ 4096
#define L_SZ 100
#define N_SZ 50
#define E_SZ 64

// ---------------- WMMA fragment loaders (wave32, 16x16x32 f16) ----------------
// A: 16x32 f16, row-major LDS tile. lane m = lane&15, hi = lane>>4.
// elems 0..7  -> K = k0 + hi*8 + r
// elems 8..15 -> K = k0 + 16 + hi*8 + r
__device__ __forceinline__ v16h lds_load_A(const h16* A, int ldA, int m0, int k0, int lane) {
    int m  = m0 + (lane & 15);
    int hi = lane >> 4;
    const h16* p = A + (size_t)m * ldA + k0 + hi * 8;
    v16h a;
#pragma unroll
    for (int r = 0; r < 8; ++r) { a[r] = p[r]; a[8 + r] = p[16 + r]; }
    return a;
}

// B: 32x16 f16 from W^T stored [N][K]. lane n = lane&15, hi = lane>>4.
// elems 0..7 -> K = k0 + hi*8 + e, elems 8..15 -> K = k0 + 16 + hi*8 + e
__device__ __forceinline__ v16h lds_load_B(const h16* WT, int K, int n0, int k0, int lane) {
    int n  = n0 + (lane & 15);
    int hi = lane >> 4;
    const h16* p = WT + (size_t)n * K + k0 + hi * 8;
    v16h b;
#pragma unroll
    for (int e = 0; e < 8; ++e) { b[e] = p[e]; b[8 + e] = p[16 + e]; }
    return b;
}

// One GEMM stage: H[R x 64] = act(A[R x K] @ W[K x 64] + bias)
// 4 waves; wave wv owns tiles t = wv, wv+4, ... of MT*4 (m-tile, n-tile) pairs.
// C/D layout: elem r -> M = m0 + r + 8*hi, N = n0 + (lane&15)
__device__ __forceinline__ void gemm_stage(const h16* A, int ldA, const h16* WT, int K,
                                           const float* bias, bool relu,
                                           h16* H, int ldH, int MT, int lane, int wv) {
    const int ntiles = MT * 4;
    for (int t = wv; t < ntiles; t += 4) {
        const int m0 = (t >> 2) << 4;
        const int n0 = (t & 3) << 4;
        v8f acc = {};
        for (int k0 = 0; k0 < K; k0 += 32) {
            v16h a  = lds_load_A(A, ldA, m0, k0, lane);
            v16h bb = lds_load_B(WT, K, n0, k0, lane);
            acc = __builtin_amdgcn_wmma_f32_16x16x32_f16(false, a, false, bb,
                                                         (short)0, acc, false, false);
        }
        const int n  = n0 + (lane & 15);
        const int hi = lane >> 4;
        const float bv = bias[n];
#pragma unroll
        for (int r = 0; r < 8; ++r) {
            float v = acc[r] + bv;
            if (relu) v = fmaxf(v, 0.0f);
            H[(size_t)(m0 + r + 8 * hi) * ldH + n] = (h16)v;
        }
    }
}

// ---------------- weight prep: f32 [K][N] -> f16 [N][K] ----------------
__global__ void transpose_f16_kernel(const float* __restrict__ src, h16* __restrict__ dst,
                                     int K, int N) {
    int i = blockIdx.x * blockDim.x + threadIdx.x;
    if (i < K * N) {
        int k = i / N, n = i - k * N;
        dst[(size_t)n * K + k] = (h16)src[i];
    }
}

// ---------------- attention pooling ----------------
// one block (128 thr = 4 waves) per batch row
__global__ __launch_bounds__(128)
void attn_pool_kernel(const int* __restrict__ idx, int T, int Lstride,
                      const float* __restrict__ tab,
                      const int* __restrict__ uid, const float* __restrict__ user_tab,
                      const h16* __restrict__ W0T, const float* __restrict__ b0,
                      const h16* __restrict__ W1T, const float* __restrict__ b1,
                      const float* __restrict__ w2, const float* __restrict__ b2,
                      h16* __restrict__ feat_out) {
    const int b    = blockIdx.x;
    const int tid  = threadIdx.x;
    const int lane = tid & 31;
    const int wv   = tid >> 5;
    const int MT   = (T + 15) >> 4;
    const int R    = MT << 4;          // padded rows (<=112)

    __shared__ h16  sCat[112 * 128];
    __shared__ h16  sH1[112 * 64];
    __shared__ h16  sH2[112 * 64];
    __shared__ h16  sW0[64 * 128];
    __shared__ h16  sW1[64 * 64];
    __shared__ float sS[112];
    __shared__ float sRed[128];

    for (int i = tid; i < 64 * 128; i += 128) sW0[i] = W0T[i];
    for (int i = tid; i < 64 * 64;  i += 128) sW1[i] = W1T[i];

    const float* ur = user_tab + (size_t)uid[b] * E_SZ;
    for (int i = tid; i < R * 64; i += 128) {
        int t = i >> 6, e = i & 63;
        float iv = 0.0f;
        if (t < T) iv = tab[(size_t)idx[(size_t)b * Lstride + t] * E_SZ + e];
        sCat[t * 128 + e]      = (h16)iv;
        sCat[t * 128 + 64 + e] = (t < T) ? (h16)ur[e] : (h16)0.0f;
    }
    __syncthreads();

    gemm_stage(sCat, 128, sW0, 128, b0, true, sH1, 64, MT, lane, wv);
    __syncthreads();
    gemm_stage(sH1, 64, sW1, 64, b1, true, sH2, 64, MT, lane, wv);
    __syncthreads();

    // scores + mask
    if (tid < R) {
        float s = -3.0e38f;
        if (tid < T) {
            float acc = b2[0];
            for (int e = 0; e < 64; ++e) acc += (float)sH2[tid * 64 + e] * w2[e];
            if (idx[(size_t)b * Lstride + tid] == 0) acc -= 1e9f;
            s = acc;
        }
        sS[tid] = s;
    }
    __syncthreads();

    // softmax over rows
    sRed[tid] = (tid < R) ? sS[tid] : -3.0e38f;
    __syncthreads();
    for (int off = 64; off > 0; off >>= 1) {
        if (tid < off) sRed[tid] = fmaxf(sRed[tid], sRed[tid + off]);
        __syncthreads();
    }
    const float mx = sRed[0];
    __syncthreads();
    const float ex = (tid < T) ? expf(sS[tid] - mx) : 0.0f;
    sRed[tid] = ex;
    __syncthreads();
    for (int off = 64; off > 0; off >>= 1) {
        if (tid < off) sRed[tid] += sRed[tid + off];
        __syncthreads();
    }
    const float inv = 1.0f / sRed[0];
    __syncthreads();
    if (tid < R) sS[tid] = (tid < T) ? ex * inv : 0.0f;
    __syncthreads();

    // pooled = sum_t a[t] * items[t]
    if (tid < 64) {
        float acc = 0.0f;
        for (int t = 0; t < T; ++t) acc += sS[t] * (float)sCat[t * 128 + tid];
        feat_out[(size_t)b * E_SZ + tid] = (h16)acc;
    }
}

// ---------------- hu = concat(seq,nbr)@sn + user MLP ----------------
__global__ __launch_bounds__(128)
void hu_kernel(const h16* __restrict__ seq_feat, const h16* __restrict__ nbr_feat,
               const h16* __restrict__ snT, const float* __restrict__ snb,
               const h16* __restrict__ um1T, const float* __restrict__ umb1,
               const h16* __restrict__ um2T, const float* __restrict__ umb2,
               h16* __restrict__ hu2) {
    const int row0 = blockIdx.x * 64;
    const int tid  = threadIdx.x;
    const int lane = tid & 31;
    const int wv   = tid >> 5;

    __shared__ h16 sCat[64 * 128];
    __shared__ h16 sA[64 * 64];
    __shared__ h16 sB[64 * 64];
    __shared__ h16 sSN[64 * 128];
    __shared__ h16 sU1[64 * 64];
    __shared__ h16 sU2[64 * 64];

    for (int i = tid; i < 64 * 128; i += 128) sSN[i] = snT[i];
    for (int i = tid; i < 64 * 64;  i += 128) { sU1[i] = um1T[i]; sU2[i] = um2T[i]; }
    for (int i = tid; i < 64 * 64; i += 128) {
        int r = i >> 6, e = i & 63;
        sCat[r * 128 + e]      = seq_feat[(size_t)(row0 + r) * E_SZ + e];
        sCat[r * 128 + 64 + e] = nbr_feat[(size_t)(row0 + r) * E_SZ + e];
    }
    __syncthreads();

    gemm_stage(sCat, 128, sSN, 128, snb, false, sA, 64, 4, lane, wv);   // hu
    __syncthreads();
    gemm_stage(sA, 64, sU1, 64, umb1, true, sB, 64, 4, lane, wv);       // relu(hu@um1+b)
    __syncthreads();
    gemm_stage(sB, 64, sU2, 64, umb2, false, hu2 + (size_t)row0 * E_SZ, 64, 4, lane, wv);
}

// ---------------- prediction head: one block per (batch, pos|neg) ----------------
__global__ __launch_bounds__(128)
void pred_kernel(const int* __restrict__ idx, const float* __restrict__ item_tab,
                 const h16* __restrict__ hu2,
                 const h16* __restrict__ im1T, const float* __restrict__ imb1,
                 const h16* __restrict__ im2T, const float* __restrict__ imb2,
                 const h16* __restrict__ pm1T, const float* __restrict__ pmb1,
                 const h16* __restrict__ pm2T, const float* __restrict__ pmb2,
                 const float* __restrict__ w3, const float* __restrict__ b3,
                 float* __restrict__ out) {
    const int b    = blockIdx.x;
    const int tid  = threadIdx.x;
    const int lane = tid & 31;
    const int wv   = tid >> 5;
    const int T = L_SZ, MT = 7, R = 112;

    __shared__ h16 sT1[112 * 64];
    __shared__ h16 sT2[112 * 64];
    __shared__ h16 sCat[112 * 128];
    __shared__ h16 sW1[64 * 64];
    __shared__ h16 sW2[64 * 64];
    __shared__ h16 sP1[64 * 128];
    __shared__ h16 sP2[64 * 64];

    for (int i = tid; i < 64 * 64;  i += 128) { sW1[i] = im1T[i]; sW2[i] = im2T[i]; sP2[i] = pm2T[i]; }
    for (int i = tid; i < 64 * 128; i += 128) sP1[i] = pm1T[i];

    const h16* hu = hu2 + (size_t)b * E_SZ;
    for (int i = tid; i < R * 64; i += 128) {
        int t = i >> 6, e = i & 63;
        float iv = 0.0f;
        if (t < T) iv = item_tab[(size_t)idx[(size_t)b * L_SZ + t] * E_SZ + e];
        sT1[i] = (h16)iv;
        sCat[t * 128 + e] = (t < T) ? hu[e] : (h16)0.0f;   // left half = hu'
    }
    __syncthreads();

    gemm_stage(sT1, 64, sW1, 64, imb1, true, sT2, 64, MT, lane, wv);     // relu(item@im1+b)
    __syncthreads();
    gemm_stage(sT2, 64, sW2, 64, imb2, false, sCat + 64, 128, MT, lane, wv); // right half = hi
    __syncthreads();
    gemm_stage(sCat, 128, sP1, 128, pmb1, true, sT1, 64, MT, lane, wv);  // relu(x@pm1+b)
    __syncthreads();
    gemm_stage(sT1, 64, sP2, 64, pmb2, true, sT2, 64, MT, lane, wv);     // relu(@pm2+b)
    __syncthreads();

    if (tid < T) {
        float acc = b3[0];
        for (int e = 0; e < 64; ++e) acc += (float)sT2[tid * 64 + e] * w3[e];
        out[(size_t)b * L_SZ + tid] = acc;
    }
}

// ---------------- host launch ----------------
extern "C" void kernel_launch(void* const* d_in, const int* in_sizes, int n_in,
                              void* d_out, int out_size, void* d_ws, size_t ws_size,
                              hipStream_t stream) {
    (void)in_sizes; (void)n_in; (void)out_size; (void)ws_size;

    const int*   uid       = (const int*)d_in[0];
    const int*   seq       = (const int*)d_in[1];
    const int*   pos       = (const int*)d_in[2];
    const int*   neg       = (const int*)d_in[3];
    const int*   nbr       = (const int*)d_in[4];
    const float* user_embs = (const float*)d_in[5];
    const float* item_embs = (const float*)d_in[6];
    const float* ia0_w = (const float*)d_in[7];  const float* ia0_b = (const float*)d_in[8];
    const float* ia1_w = (const float*)d_in[9];  const float* ia1_b = (const float*)d_in[10];
    const float* ia2_w = (const float*)d_in[11]; const float* ia2_b = (const float*)d_in[12];
    const float* ua0_w = (const float*)d_in[13]; const float* ua0_b = (const float*)d_in[14];
    const float* ua1_w = (const float*)d_in[15]; const float* ua1_b = (const float*)d_in[16];
    const float* ua2_w = (const float*)d_in[17]; const float* ua2_b = (const float*)d_in[18];
    const float* sn_w  = (const float*)d_in[19]; const float* sn_b  = (const float*)d_in[20];
    const float* um_w1 = (const float*)d_in[21]; const float* um_b1 = (const float*)d_in[22];
    const float* um_w2 = (const float*)d_in[23]; const float* um_b2 = (const float*)d_in[24];
    const float* im_w1 = (const float*)d_in[25]; const float* im_b1 = (const float*)d_in[26];
    const float* im_w2 = (const float*)d_in[27]; const float* im_b2 = (const float*)d_in[28];
    const float* pm_w1 = (const float*)d_in[29]; const float* pm_b1 = (const float*)d_in[30];
    const float* pm_w2 = (const float*)d_in[31]; const float* pm_b2 = (const float*)d_in[32];
    const float* pm_w3 = (const float*)d_in[33]; const float* pm_b3 = (const float*)d_in[34];

    // workspace layout (halfs)
    h16* ws = (h16*)d_ws;
    size_t off = 0;
    h16* ia0T = ws + off; off += 128 * 64;
    h16* ia1T = ws + off; off +=  64 * 64;
    h16* ua0T = ws + off; off += 128 * 64;
    h16* ua1T = ws + off; off +=  64 * 64;
    h16* snT  = ws + off; off += 128 * 64;
    h16* um1T = ws + off; off +=  64 * 64;
    h16* um2T = ws + off; off +=  64 * 64;
    h16* im1T = ws + off; off +=  64 * 64;
    h16* im2T = ws + off; off +=  64 * 64;
    h16* pm1T = ws + off; off += 128 * 64;
    h16* pm2T = ws + off; off +=  64 * 64;
    h16* seq_feat = ws + off; off += (size_t)B_SZ * E_SZ;
    h16* nbr_feat = ws + off; off += (size_t)B_SZ * E_SZ;
    h16* hu2      = ws + off; off += (size_t)B_SZ * E_SZ;

    auto xpose = [&](const float* src, h16* dst, int K, int N) {
        int n = K * N;
        transpose_f16_kernel<<<(n + 255) / 256, 256, 0, stream>>>(src, dst, K, N);
    };
    xpose(ia0_w, ia0T, 128, 64);
    xpose(ia1_w, ia1T,  64, 64);
    xpose(ua0_w, ua0T, 128, 64);
    xpose(ua1_w, ua1T,  64, 64);
    xpose(sn_w,  snT,  128, 64);
    xpose(um_w1, um1T,  64, 64);
    xpose(um_w2, um2T,  64, 64);
    xpose(im_w1, im1T,  64, 64);
    xpose(im_w2, im2T,  64, 64);
    xpose(pm_w1, pm1T, 128, 64);
    xpose(pm_w2, pm2T,  64, 64);

    attn_pool_kernel<<<B_SZ, 128, 0, stream>>>(seq, L_SZ, L_SZ, item_embs, uid, user_embs,
                                               ia0T, ia0_b, ia1T, ia1_b, ia2_w, ia2_b, seq_feat);
    attn_pool_kernel<<<B_SZ, 128, 0, stream>>>(nbr, N_SZ, N_SZ, user_embs, uid, user_embs,
                                               ua0T, ua0_b, ua1T, ua1_b, ua2_w, ua2_b, nbr_feat);
    hu_kernel<<<B_SZ / 64, 128, 0, stream>>>(seq_feat, nbr_feat, snT, sn_b,
                                             um1T, um_b1, um2T, um_b2, hu2);

    float* out = (float*)d_out;
    pred_kernel<<<B_SZ, 128, 0, stream>>>(pos, item_embs, hu2, im1T, im_b1, im2T, im_b2,
                                          pm1T, pm_b1, pm2T, pm_b2, pm_w3, pm_b3, out);
    pred_kernel<<<B_SZ, 128, 0, stream>>>(neg, item_embs, hu2, im1T, im_b1, im2T, im_b2,
                                          pm1T, pm_b1, pm2T, pm_b2, pm_w3, pm_b3,
                                          out + (size_t)B_SZ * L_SZ);
}
